// GAT_20564303414151
// MI455X (gfx1250) — matrix operand order repaired and checked by reading, not compiled, not run on hardware
//
#include <hip/hip_runtime.h>
#include <cstdint>

// ---------------------------------------------------------------------------
// GAT layer for MI455X (gfx1250): B=8, N=2048, D=128.
//   h' = h @ W + b            (f32 WMMA 16x16x4, exact)
//   s_i = h'.w_src + b_att ; t_j = h'.w_dst
//   w_ij = exp(relu(s_i + t_j)) * a_ij ; out = (w/rowsum) @ h' + h'
// Big matmul: bf16 WMMA 16x16x32 with hi/lo 3-term split (~f32 accuracy).
// Tiles staged LDS-side with the Tensor Data Mover (tensor_load_to_lds).
// ---------------------------------------------------------------------------

typedef float  v2f   __attribute__((ext_vector_type(2)));
typedef float  v8f   __attribute__((ext_vector_type(8)));
typedef __bf16 v16bf __attribute__((ext_vector_type(16)));
typedef unsigned int u32x4 __attribute__((ext_vector_type(4)));
typedef int          i32x4 __attribute__((ext_vector_type(4)));
typedef int          i32x8 __attribute__((ext_vector_type(8)));

#define GAT_B 8
#define GAT_N 2048
#define GAT_D 128

// ---------------------------------------------------------------------------
// Kernel 1: projection h' = h @ W + b, via V_WMMA_F32_16X16X4_F32.
// Also emits bf16 hi/lo copies in d-major layout hT[b][d][j] for the fused
// kernel's B-operand staging.
// ---------------------------------------------------------------------------
__global__ __launch_bounds__(256) void gat_proj_kernel(
    const float* __restrict__ h, const float* __restrict__ W,
    const float* __restrict__ bias,
    float* __restrict__ hprime,
    __bf16* __restrict__ hhiT, __bf16* __restrict__ hloT)
{
  const int lane = threadIdx.x & 31;
  const int wave = threadIdx.x >> 5;
  const int tile = blockIdx.x * 8 + wave;   // 8192 (16-row x 16-col) tiles
  const int ct = tile & 7;
  const int rt = tile >> 3;
  const int i0 = rt * 16;
  const int c0 = ct * 16;
  const int m  = lane & 15;
  const int hi = lane >> 4;

  v8f acc = {0.f, 0.f, 0.f, 0.f, 0.f, 0.f, 0.f, 0.f};

  const float* hrow = h + (size_t)(i0 + m) * GAT_D;
  #pragma unroll 4
  for (int k0 = 0; k0 < GAT_D; k0 += 4) {
    // A (16x4 f32): V0 = K {0|2}, V1 = K {1|3} per lane half.
    const float2 av = *(const float2*)(hrow + k0 + hi * 2);
    v2f A = {av.x, av.y};
    // B (4x16 f32): V0 = row K {0|2}, V1 = row K {1|3}, col = c0 + m.
    const int kB = k0 + hi * 2;
    v2f Bv = {W[(size_t)kB * GAT_D + c0 + m],
              W[(size_t)(kB + 1) * GAT_D + c0 + m]};
    acc = __builtin_amdgcn_wmma_f32_16x16x4_f32(false, A, false, Bv,
                                                (short)0, acc, false, false);
  }

  const float bb = bias[c0 + m];
  #pragma unroll
  for (int r = 0; r < 8; ++r) {
    const int row = i0 + r + hi * 8;            // C/D layout: M = r + 8*hi
    const int c   = c0 + m;
    const float x = acc[r] + bb;
    hprime[(size_t)row * GAT_D + c] = x;
    const __bf16 xh = (__bf16)x;
    const __bf16 xl = (__bf16)(x - (float)xh);
    const int bidx = row >> 11;                 // / N
    const int iloc = row & (GAT_N - 1);
    const size_t tIdx = ((size_t)bidx * GAT_D + c) * GAT_N + iloc;
    hhiT[tIdx] = xh;
    hloT[tIdx] = xl;
  }
}

// ---------------------------------------------------------------------------
// Kernel 2: s_i = h'_i . w_att[:D] + b_att ; t_j = h'_j . w_att[D:]
// One wave per row, shuffle reduction.
// ---------------------------------------------------------------------------
__global__ __launch_bounds__(256) void gat_st_kernel(
    const float* __restrict__ hprime, const float* __restrict__ w_att,
    const float* __restrict__ b_att,
    float* __restrict__ s, float* __restrict__ t)
{
  const int lane = threadIdx.x & 31;
  const int row  = (int)((blockIdx.x * blockDim.x + threadIdx.x) >> 5);
  const float* hr = hprime + (size_t)row * GAT_D;
  float ss = 0.f, tt = 0.f;
  #pragma unroll
  for (int c = lane; c < GAT_D; c += 32) {
    const float v = hr[c];
    ss = fmaf(v, w_att[c], ss);
    tt = fmaf(v, w_att[GAT_D + c], tt);
  }
  #pragma unroll
  for (int off = 16; off; off >>= 1) {
    ss += __shfl_xor(ss, off, 32);
    tt += __shfl_xor(tt, off, 32);
  }
  if (lane == 0) { s[row] = ss + b_att[0]; t[row] = tt; }
}

// ---------------------------------------------------------------------------
// TDM descriptor builders (CDNA5 ISA ch.8).  Group0 = 4 dwords, Group1 = 8.
// Only group0 words 2..3 (global_addr) change per j-step.
// ---------------------------------------------------------------------------
static __device__ inline u32x4 tdm_group0(unsigned lds_addr,
                                          unsigned long long ga) {
  u32x4 g;
  g[0] = 1u;                                   // count=1, user descriptor
  g[1] = lds_addr;                             // lds_addr [63:32]
  g[2] = (unsigned)ga;                         // global_addr [95:64]
  g[3] = (unsigned)((ga >> 32) & 0x1FFFFFFu)   // global_addr [120:96]
       | (2u << 30);                           // type=2 ("image")
  return g;
}

static __device__ inline i32x8 tdm_group1(unsigned data_size_log2,
                                          unsigned tensor_d0, unsigned tensor_d1,
                                          unsigned tile_d0, unsigned tile_d1,
                                          unsigned long long stride0_elems) {
  i32x8 g;
  g[0] = (int)(data_size_log2 << 16);          // workgroup_mask=0, data_size
  g[1] = (int)((tensor_d0 & 0xFFFFu) << 16);   // tensor_dim0 [79:48]
  g[2] = (int)(((tensor_d0 >> 16) & 0xFFFFu) | ((tensor_d1 & 0xFFFFu) << 16));
  g[3] = (int)(((tensor_d1 >> 16) & 0xFFFFu) | (tile_d0 << 16));
  g[4] = (int)(tile_d1 & 0xFFFFu);             // tile_dim1; tile_dim2=0
  g[5] = (int)(unsigned)(stride0_elems & 0xFFFFFFFFull); // dim0_stride lo
  g[6] = (int)(unsigned)((stride0_elems >> 32) & 0xFFFFull);
  g[7] = 0;
  return g;
}

// ---------------------------------------------------------------------------
// Kernel 3: fused attention + matmul.
// Block = 4 waves = 64 i-rows.  K loop over j in steps of 32.
// ---------------------------------------------------------------------------
__global__ __launch_bounds__(128) void gat_fused_kernel(
    const float* __restrict__ a, const float* __restrict__ hprime,
    const __bf16* __restrict__ hhiT, const __bf16* __restrict__ hloT,
    const float* __restrict__ s, const float* __restrict__ t,
    float* __restrict__ out)
{
  __shared__ float  a_tile[64 * 32];     // 8 KB   a[b, i0..i0+63, j0..j0+31]
  __shared__ __bf16 hhi_tile[128 * 32];  // 8 KB   h_hiT[b, 0..127, j0..j0+31]
  __shared__ __bf16 hlo_tile[128 * 32];  // 8 KB
  __shared__ float  t_tile[32];

  const int lane = threadIdx.x & 31;
  const int wave = threadIdx.x >> 5;
  const int b    = blockIdx.x >> 5;     // 32 row-blocks per batch
  const int it   = blockIdx.x & 31;
  const int iB0  = it * 64;
  const int i0   = iB0 + wave * 16;
  const int m    = lane & 15;
  const int hi   = lane >> 4;

  const float s_i = s[b * GAT_N + i0 + m];

  v8f acc[8];
  #pragma unroll
  for (int dt = 0; dt < 8; ++dt)
    acc[dt] = (v8f){0.f, 0.f, 0.f, 0.f, 0.f, 0.f, 0.f, 0.f};
  float rowsum = 0.f;

  // Static descriptor parts (wave-uniform -> SGPRs).
  const i32x8 g1_a = tdm_group1(/*4B*/2, GAT_N, GAT_N, 32, 64, GAT_N);
  const i32x8 g1_h = tdm_group1(/*2B*/1, GAT_N, GAT_D, 32, GAT_D, GAT_N);
  const i32x8 g1_t = tdm_group1(/*4B*/2, GAT_N, 1, 32, 1, GAT_N);
  const i32x4 gz4 = {0, 0, 0, 0};
  const i32x8 gz8 = {0, 0, 0, 0, 0, 0, 0, 0};

  const unsigned lds_a   = (unsigned)(uintptr_t)&a_tile[0];
  const unsigned lds_hhi = (unsigned)(uintptr_t)&hhi_tile[0];
  const unsigned lds_hlo = (unsigned)(uintptr_t)&hlo_tile[0];
  const unsigned lds_t   = (unsigned)(uintptr_t)&t_tile[0];

  const unsigned long long gaA0 =
      (unsigned long long)(uintptr_t)(a + ((size_t)b * GAT_N + iB0) * GAT_N);
  const unsigned long long gaH0 =
      (unsigned long long)(uintptr_t)(hhiT + (size_t)b * GAT_D * GAT_N);
  const unsigned long long gaL0 =
      (unsigned long long)(uintptr_t)(hloT + (size_t)b * GAT_D * GAT_N);
  const unsigned long long gaT0 =
      (unsigned long long)(uintptr_t)(t + (size_t)b * GAT_N);

  for (int jt = 0; jt < GAT_N / 32; ++jt) {
    const int j0 = jt * 32;
    __syncthreads();  // previous tile fully consumed before DMA overwrite
    if (wave == 0) {
      __builtin_amdgcn_tensor_load_to_lds(
          tdm_group0(lds_a, gaA0 + (size_t)j0 * 4), g1_a, gz4, gz4, gz8, 0);
    } else if (wave == 1) {
      __builtin_amdgcn_tensor_load_to_lds(
          tdm_group0(lds_hhi, gaH0 + (size_t)j0 * 2), g1_h, gz4, gz4, gz8, 0);
    } else if (wave == 2) {
      __builtin_amdgcn_tensor_load_to_lds(
          tdm_group0(lds_hlo, gaL0 + (size_t)j0 * 2), g1_h, gz4, gz4, gz8, 0);
    } else {
      __builtin_amdgcn_tensor_load_to_lds(
          tdm_group0(lds_t, gaT0 + (size_t)j0 * 4), g1_t, gz4, gz4, gz8, 0);
    }
    __builtin_amdgcn_s_wait_tensorcnt(0);
    __syncthreads();  // DMA'd tiles visible to all waves

    // ---- A fragment: w = exp(relu(s_i + t_j)) * a_ij as bf16 hi + lo.
    // A 16x32, contiguous k order chosen here (B matches): lane half selects K+16; V0..7 = K 0..15
    // (each VGPR packs K, K+1).
    union { v16bf v; __bf16 e[16]; } awhi, awlo;
    const int rowLoc = wave * 16 + m;
    #pragma unroll
    for (int v = 0; v < 8; ++v) {
      const int K = 2 * v + hi * 16;
      const float2 ap = *(const float2*)&a_tile[rowLoc * 32 + K];
      const float2 tp = *(const float2*)&t_tile[K];
      float e0 = s_i + tp.x; e0 = e0 > 0.f ? e0 : 0.f;
      float e1 = s_i + tp.y; e1 = e1 > 0.f ? e1 : 0.f;
      const float w0 = __expf(e0) * ap.x;
      const float w1 = __expf(e1) * ap.y;
      rowsum += w0 + w1;
      const __bf16 w0h = (__bf16)w0;
      const __bf16 w1h = (__bf16)w1;
      awhi.e[2 * v]     = w0h;
      awhi.e[2 * v + 1] = w1h;
      awlo.e[2 * v]     = (__bf16)(w0 - (float)w0h);
      awlo.e[2 * v + 1] = (__bf16)(w1 - (float)w1h);
    }

    // ---- 8 d-tiles x 3 hi/lo terms of V_WMMA_F32_16X16X32_BF16.
    // B 32x16, same contiguous k order as A: lane col = lane&15, K half = (lane>>4)*16,
    // VGPR v packs rows K = base+2v, base+2v+1 -> one b32 LDS load.
    #pragma unroll
    for (int dt = 0; dt < 8; ++dt) {
      union { v16bf v; unsigned u[8]; } bhi, blo;
      const int d = dt * 16 + m;
      #pragma unroll
      for (int v = 0; v < 8; ++v) {
        const int jj = hi * 16 + 2 * v;
        bhi.u[v] = *(const unsigned*)&hhi_tile[d * 32 + jj];
        blo.u[v] = *(const unsigned*)&hlo_tile[d * 32 + jj];
      }
      acc[dt] = __builtin_amdgcn_wmma_f32_16x16x32_bf16(
          false, awhi.v, false, bhi.v, (short)0, acc[dt], false, false);
      acc[dt] = __builtin_amdgcn_wmma_f32_16x16x32_bf16(
          false, awlo.v, false, bhi.v, (short)0, acc[dt], false, false);
      acc[dt] = __builtin_amdgcn_wmma_f32_16x16x32_bf16(
          false, awhi.v, false, blo.v, (short)0, acc[dt], false, false);
    }
  }

  // Row m's total = lane m partial + lane m+16 partial.
  rowsum += __shfl_xor(rowsum, 16, 32);

  // Epilogue: D element (VGPR r, lane) is row M = r + 8*hi, col dt*16 + m.
  #pragma unroll
  for (int r = 0; r < 8; ++r) {
    const int rr = r + hi * 8;
    const float rs = __shfl(rowsum, rr, 32);   // bpermute broadcast
    const float inv = 1.0f / rs;
    const int i = i0 + rr;
    #pragma unroll
    for (int dt = 0; dt < 8; ++dt) {
      const int c = dt * 16 + m;
      const size_t idx = ((size_t)(b * GAT_N + i)) * GAT_D + c;
      out[idx] = acc[dt][r] * inv + hprime[idx];
    }
  }
}

// ---------------------------------------------------------------------------
extern "C" void kernel_launch(void* const* d_in, const int* in_sizes, int n_in,
                              void* d_out, int out_size, void* d_ws,
                              size_t ws_size, hipStream_t stream) {
  (void)in_sizes; (void)n_in; (void)out_size; (void)ws_size;
  const float* a      = (const float*)d_in[0];
  const float* h      = (const float*)d_in[1];
  const float* W_proj = (const float*)d_in[2];
  const float* b_proj = (const float*)d_in[3];
  const float* w_att  = (const float*)d_in[4];
  const float* b_att  = (const float*)d_in[5];
  float* out = (float*)d_out;

  char* ws = (char*)d_ws;
  float*  hprime = (float*)(ws);                       // 8 MiB f32 [B*N][D]
  __bf16* hhiT   = (__bf16*)(ws + (8u << 20));         // 4 MiB bf16 [B][D][N]
  __bf16* hloT   = (__bf16*)(ws + (12u << 20));        // 4 MiB bf16 [B][D][N]
  float*  s      = (float*)(ws + (16u << 20));         // 64 KiB
  float*  t      = (float*)(ws + (16u << 20) + (64u << 10));

  // 1) projection (f32 WMMA): 8192 wave-tiles / 8 waves per block
  hipLaunchKernelGGL(gat_proj_kernel, dim3(1024), dim3(256), 0, stream,
                     h, W_proj, b_proj, hprime, hhiT, hloT);
  // 2) s/t scores: one wave per row (B*N = 16384 rows)
  hipLaunchKernelGGL(gat_st_kernel, dim3(2048), dim3(256), 0, stream,
                     hprime, w_att, b_att, s, t);
  // 3) fused attention + bf16-WMMA matmul + residual
  hipLaunchKernelGGL(gat_fused_kernel, dim3(GAT_B * (GAT_N / 64)), dim3(128),
                     0, stream, a, hprime, hhiT, hloT, s, t, out);
}
